// SimpleModuleNet_87677462380567
// MI455X (gfx1250) — matrix operand order repaired and checked by reading, not compiled, not run on hardware
//
#include <hip/hip_runtime.h>
#include <hip/hip_bf16.h>
#include <stdint.h>

typedef __attribute__((ext_vector_type(16))) _Float16 v16h;
typedef __attribute__((ext_vector_type(8)))  float    v8f;
typedef __attribute__((ext_vector_type(4)))  unsigned int u32x4;
typedef __attribute__((ext_vector_type(8)))  int      i32x8;
typedef __attribute__((ext_vector_type(4)))  int      i32x4;

#define Bsz 256

// ---------------------------------------------------------------------------
// A-tile LDS layout: identity within each 32-wide K group, matching the
// contiguous B fragment order: [k0..15 | k16..31]. Bit5 (WMMA k-step) preserved.
// ---------------------------------------------------------------------------
__device__ __forceinline__ int permA(int k) {
  return k;
}

// Contiguous 16-half (32B) LDS fragment load -> ds_load_b128 pair.
__device__ __forceinline__ v16h frag16(const _Float16* p) {
  return *reinterpret_cast<const v16h*>(p);
}

// ---------------------------------------------------------------------------
// Tensor Data Mover: load a 2D tile of f16 (tile1 rows x tile0 elems, row
// stride stride0 elems) from global into LDS at lds_addr. D# per ISA 8.3/8.4:
//   group0: count=1 | lds_addr | global_addr[56:0] | type=2
//   group1: data_size=1(2B) | tensor_dim0/1 | tile_dim0/1 | dim0_stride
// Issued once per wave (EXEC ignored); tracked by TENSORcnt.
// ---------------------------------------------------------------------------
__device__ __forceinline__ void tdm_load_2d_f16(uint32_t lds_addr,
                                                const void* gaddr,
                                                uint32_t dim0, uint32_t dim1,
                                                uint32_t stride0,
                                                uint32_t tile0, uint32_t tile1) {
#if __has_builtin(__builtin_amdgcn_tensor_load_to_lds)
  const uint64_t ga = (uint64_t)(uintptr_t)gaddr;
  u32x4 g0;
  g0[0] = 1u;                                   // count=1 (user descriptor)
  g0[1] = lds_addr;                             // lds_addr bits 63:32
  g0[2] = (uint32_t)ga;                         // global_addr[31:0]
  g0[3] = (uint32_t)((ga >> 32) & 0x01FFFFFFu)  // global_addr[56:32]
          | (2u << 30);                         // type=2 ("image")
  i32x8 g1;
  g1[0] = 0x10000;                              // data_size=1 (2 bytes)
  g1[1] = (int)(dim0 << 16);                    // tensor_dim0[15:0] @63:48
  g1[2] = (int)((dim0 >> 16) | (dim1 << 16));   // dim0[31:16] | dim1[15:0]
  g1[3] = (int)((dim1 >> 16) | (tile0 << 16));  // dim1[31:16] | tile_dim0
  g1[4] = (int)(tile1 & 0xFFFF);                // tile_dim1 (tile_dim2=0)
  g1[5] = (int)stride0;                         // dim0_stride[31:0]
  g1[6] = 0;                                    // dim0_stride[47:32]|dim1_strd
  g1[7] = 0;
  i32x4 z4 = {0, 0, 0, 0};
#if defined(__clang_major__) && __clang_major__ >= 23
  i32x8 z8 = {0, 0, 0, 0, 0, 0, 0, 0};
  __builtin_amdgcn_tensor_load_to_lds(g0, g1, z4, z4, z8, 0);
#else
  __builtin_amdgcn_tensor_load_to_lds(g0, g1, z4, z4, 0);
#endif
#endif
}

__device__ __forceinline__ void tdm_wait() {
#if __has_builtin(__builtin_amdgcn_s_wait_tensorcnt)
  __builtin_amdgcn_s_wait_tensorcnt(0);
#endif
}

// ---------------------------------------------------------------------------
// Weight prepack: fp32 [set][oc][ic][3][3]  ->  f16 [set][tap][oc][ic]
// ---------------------------------------------------------------------------
__global__ void pack_conv_w_kernel(const float* __restrict__ src,
                                   _Float16* __restrict__ dst, int total) {
  int i = blockIdx.x * blockDim.x + threadIdx.x;
  if (i >= total) return;
  int t = i;
  const int ic = t & 63;  t >>= 6;
  const int oc = t & 63;  t >>= 6;
  const int tap = t % 9;
  const int set = t / 9;
  dst[i] = (_Float16)src[(((size_t)(set * 64 + oc) * 64 + ic) * 9) + tap];
}

// Plain fp32 -> f16 convert (same layout). Used for proj_w and fc1_w.
__global__ void cvt_f16_kernel(const float* __restrict__ src,
                               _Float16* __restrict__ dst, int total) {
  int i = blockIdx.x * blockDim.x + threadIdx.x;
  if (i < total) dst[i] = (_Float16)src[i];
}

// ---------------------------------------------------------------------------
// Stem conv0: 3->64, 3x3, pad 1, 64x64, bias+relu. K=27 -> plain VALU.
// ---------------------------------------------------------------------------
__global__ void stem_conv0_kernel(const float* __restrict__ img,
                                  const float* __restrict__ w,
                                  const float* __restrict__ bias,
                                  float* __restrict__ out) {
  const int i = blockIdx.x * blockDim.x + threadIdx.x;
  const int x  = i & 63;
  const int y  = (i >> 6) & 63;
  const int oc = (i >> 12) & 63;
  const int b  = i >> 18;
  float acc = bias[oc];
#pragma unroll
  for (int ic = 0; ic < 3; ++ic) {
#pragma unroll
    for (int dy = 0; dy < 3; ++dy) {
      const int yy = y + dy - 1;
#pragma unroll
      for (int dx = 0; dx < 3; ++dx) {
        const int xx = x + dx - 1;
        const bool ok = (unsigned)yy < 64u && (unsigned)xx < 64u;
        const int yc = min(max(yy, 0), 63), xc = min(max(xx, 0), 63);
        float v = img[((size_t)(b * 3 + ic) << 12) + (yc << 6) + xc];
        acc += (ok ? v : 0.f) * w[((oc * 3 + ic) * 3 + dy) * 3 + dx];
      }
    }
  }
  out[i] = fmaxf(acc, 0.f);
}

// ---------------------------------------------------------------------------
// 2x2 maxpool, stride 2, NCHW.
// ---------------------------------------------------------------------------
__global__ void maxpool2_kernel(const float* __restrict__ in,
                                float* __restrict__ out,
                                int C, int Si, int total) {
  const int i = blockIdx.x * blockDim.x + threadIdx.x;
  if (i >= total) return;
  const int So = Si >> 1;
  const int x = i % So;
  int rest = i / So;
  const int y = rest % So;
  rest /= So;
  const int c = rest % C;
  const int b = rest / C;
  const size_t base = ((size_t)(b * C + c) * Si + 2 * y) * Si + 2 * x;
  float m = fmaxf(in[base], in[base + 1]);
  m = fmaxf(m, fmaxf(in[base + Si], in[base + Si + 1]));
  out[i] = m;
}

// ---------------------------------------------------------------------------
// 3x3 conv 64->64, pad 1, implicit GEMM on WMMA f16 (f32 accum).
// B (weight) tile staged by the Tensor Data Mover, overlapped with the
// VALU A-tile stage (fp32 gather + halo pad + f16 convert).
// Block 256 (8 waves), grid (S*S/64, B).
// ---------------------------------------------------------------------------
__global__ __launch_bounds__(Bsz)
void conv3x3_wmma_kernel(const float* __restrict__ in,
                         const _Float16* __restrict__ wpk,
                         const float* __restrict__ bbase,
                         const int*   __restrict__ question, int qcol,
                         const float* __restrict__ residual,
                         float* __restrict__ out,
                         int lgS, int do_relu) {
  __shared__ alignas(32) _Float16 lA[64 * 64];   // [m][k]
  __shared__ alignas(32) _Float16 lB[64 * 64];   // [n][k]

  const int S  = 1 << lgS;
  const int SS = S * S;
  const int tid  = threadIdx.x;
  const int lane = tid & 31;
  const int wave = tid >> 5;
  const int b    = blockIdx.y;
  const int mbase = blockIdx.x * 64;

  const int e = (qcol >= 0)
      ? __builtin_amdgcn_readfirstlane(question[b * 8 + qcol]) : 0;
  const _Float16* wset = wpk + (size_t)e * 9 * 64 * 64;
  const float* bias = bbase + e * 64;

  const int msub   = wave & 3;
  const int nbase2 = (wave >> 2) * 32;
  const int row = lane & 15, kh = lane >> 4;
  const uint32_t lB_addr = (uint32_t)(uintptr_t)lB;

  v8f acc0 = {};
  v8f acc1 = {};

  for (int dy = -1; dy <= 1; ++dy) {
    for (int dx = -1; dx <= 1; ++dx) {
      const int tap = (dy + 1) * 3 + (dx + 1);
      // Kick off B-tile DMA (64x64 f16, contiguous) from wave 0 only.
      if (wave == 0)
        tdm_load_2d_f16(lB_addr, wset + ((size_t)tap << 12),
                        64, 64, 64, 64, 64);
      if (tap < 8)
        __builtin_prefetch(wset + ((size_t)(tap + 1) << 12), 0, 1);

      // Stage A: 64 spatial x 64 ic, branchless zero-pad, coalesced in m.
#pragma unroll
      for (int i = tid; i < 4096; i += Bsz) {
        const int m = i & 63, j = i >> 6;
        const int pos = mbase + m;
        const int y = (pos >> lgS) + dy;
        const int x = (pos & (S - 1)) + dx;
        const bool ok = ((unsigned)y < (unsigned)S) & ((unsigned)x < (unsigned)S);
        const int yc = min(max(y, 0), S - 1);
        const int xc = min(max(x, 0), S - 1);
        const float v = in[(size_t)(b * 64 + j) * SS + (yc << lgS) + xc];
        lA[m * 64 + permA(j)] = (_Float16)(ok ? v : 0.f);
      }
      if (wave == 0) tdm_wait();
      __syncthreads();

#pragma unroll
      for (int kc = 0; kc < 2; ++kc) {
        const int ko = kc * 32 + kh * 16;
        const v16h af  = frag16(lA + (msub * 16 + row) * 64 + ko);
        const v16h bf0 = frag16(lB + (nbase2 + row) * 64 + ko);
        const v16h bf1 = frag16(lB + (nbase2 + 16 + row) * 64 + ko);
        acc0 = __builtin_amdgcn_wmma_f32_16x16x32_f16(
            false, af, false, bf0, (short)0, acc0, false, false);
        acc1 = __builtin_amdgcn_wmma_f32_16x16x32_f16(
            false, af, false, bf1, (short)0, acc1, false, false);
      }
      __syncthreads();
    }
  }

  // Epilogue: C layout lane&15 = N, VGPR r + 8*(lane>>4) = M.
#pragma unroll
  for (int r = 0; r < 8; ++r) {
    const int m = mbase + msub * 16 + r + 8 * kh;
    {
      const int ch = nbase2 + row;
      const size_t o = (size_t)(b * 64 + ch) * SS + m;
      float v = acc0[r] + bias[ch];
      if (residual) v += residual[o];
      if (do_relu) v = fmaxf(v, 0.f);
      out[o] = v;
    }
    {
      const int ch = nbase2 + 16 + row;
      const size_t o = (size_t)(b * 64 + ch) * SS + m;
      float v = acc1[r] + bias[ch];
      if (residual) v += residual[o];
      if (do_relu) v = fmaxf(v, 0.f);
      out[o] = v;
    }
  }
}

// ---------------------------------------------------------------------------
// 1x1 conv 64->512 @16x16 + bias + relu: GEMM M=B*256, N=512, K=64.
// ---------------------------------------------------------------------------
__global__ __launch_bounds__(Bsz)
void proj_wmma_kernel(const float* __restrict__ in,
                      const _Float16* __restrict__ wpk,
                      const float* __restrict__ bias,
                      float* __restrict__ out) {
  __shared__ alignas(32) _Float16 lA[64 * 64];
  __shared__ alignas(32) _Float16 lB[64 * 64];

  const int tid  = threadIdx.x;
  const int lane = tid & 31;
  const int wave = tid >> 5;
  const int mg   = blockIdx.x * 64;
  const int b    = mg >> 8;
  const int pos0 = mg & 255;
  const int nbase = blockIdx.y * 64;

  const int msub   = wave & 3;
  const int nbase2 = (wave >> 2) * 32;
  const int row = lane & 15, kh = lane >> 4;

  v8f acc0 = {};
  v8f acc1 = {};

  if (wave == 0)
    tdm_load_2d_f16((uint32_t)(uintptr_t)lB, wpk + ((size_t)nbase << 6),
                    64, 64, 64, 64, 64);
#pragma unroll
  for (int i = tid; i < 4096; i += Bsz) {
    const int m = i & 63, j = i >> 6;
    lA[m * 64 + permA(j)] =
        (_Float16)in[((size_t)(b * 64 + j) << 8) + pos0 + m];
  }
  if (wave == 0) tdm_wait();
  __syncthreads();

#pragma unroll
  for (int kc = 0; kc < 2; ++kc) {
    const int ko = kc * 32 + kh * 16;
    const v16h af  = frag16(lA + (msub * 16 + row) * 64 + ko);
    const v16h bf0 = frag16(lB + (nbase2 + row) * 64 + ko);
    const v16h bf1 = frag16(lB + (nbase2 + 16 + row) * 64 + ko);
    acc0 = __builtin_amdgcn_wmma_f32_16x16x32_f16(false, af, false, bf0,
                                                  (short)0, acc0, false, false);
    acc1 = __builtin_amdgcn_wmma_f32_16x16x32_f16(false, af, false, bf1,
                                                  (short)0, acc1, false, false);
  }
  __syncthreads();

#pragma unroll
  for (int r = 0; r < 8; ++r) {
    const int Mloc = msub * 16 + r + 8 * kh;
    {
      const int ch = nbase + nbase2 + row;
      const size_t o = ((size_t)(b * 512 + ch) << 8) + pos0 + Mloc;
      out[o] = fmaxf(acc0[r] + bias[ch], 0.f);
    }
    {
      const int ch = nbase + nbase2 + 16 + row;
      const size_t o = ((size_t)(b * 512 + ch) << 8) + pos0 + Mloc;
      out[o] = fmaxf(acc1[r] + bias[ch], 0.f);
    }
  }
}

// ---------------------------------------------------------------------------
// FC GEMM: out = relu(act[M,K] x wpk[N,K]^T + bias). wpk f16, rows stride K.
// B tile staged by TDM with row-strided descriptor.
// ---------------------------------------------------------------------------
__global__ __launch_bounds__(Bsz)
void fc_wmma_kernel(const float* __restrict__ act,
                    const _Float16* __restrict__ wpk,
                    const float* __restrict__ bias,
                    float* __restrict__ out,
                    int N, int K) {
  __shared__ alignas(32) _Float16 lA[64 * 64];
  __shared__ alignas(32) _Float16 lB[64 * 64];

  const int tid  = threadIdx.x;
  const int lane = tid & 31;
  const int wave = tid >> 5;
  const int mbase = blockIdx.x * 64;
  const int nbase = blockIdx.y * 64;

  const int msub   = wave & 3;
  const int nbase2 = (wave >> 2) * 32;
  const int row = lane & 15, kh = lane >> 4;
  const uint32_t lB_addr = (uint32_t)(uintptr_t)lB;

  v8f acc0 = {};
  v8f acc1 = {};

  for (int kb = 0; kb < K; kb += 64) {
    // B: 64 rows x 64 k f16, row stride K elems -> one TDM descriptor.
    if (wave == 0)
      tdm_load_2d_f16(lB_addr, wpk + (size_t)nbase * K + kb,
                      (uint32_t)K, (uint32_t)N, (uint32_t)K, 64, 64);
    // A: lanes consecutive in k -> coalesced along the contiguous dim.
#pragma unroll
    for (int i = tid; i < 4096; i += Bsz) {
      const int j = i & 63, m = i >> 6;
      lA[m * 64 + permA(j)] =
          (_Float16)act[(size_t)(mbase + m) * K + kb + j];
    }
    if (wave == 0) tdm_wait();
    __syncthreads();

#pragma unroll
    for (int kc = 0; kc < 2; ++kc) {
      const int ko = kc * 32 + kh * 16;
      const v16h af  = frag16(lA + (msub * 16 + row) * 64 + ko);
      const v16h bf0 = frag16(lB + (nbase2 + row) * 64 + ko);
      const v16h bf1 = frag16(lB + (nbase2 + 16 + row) * 64 + ko);
      acc0 = __builtin_amdgcn_wmma_f32_16x16x32_f16(
          false, af, false, bf0, (short)0, acc0, false, false);
      acc1 = __builtin_amdgcn_wmma_f32_16x16x32_f16(
          false, af, false, bf1, (short)0, acc1, false, false);
    }
    __syncthreads();
  }

#pragma unroll
  for (int r = 0; r < 8; ++r) {
    const int m = mbase + msub * 16 + r + 8 * kh;
    {
      const int n = nbase + nbase2 + row;
      out[(size_t)m * N + n] = fmaxf(acc0[r] + bias[n], 0.f);
    }
    {
      const int n = nbase + nbase2 + 16 + row;
      out[(size_t)m * N + n] = fmaxf(acc1[r] + bias[n], 0.f);
    }
  }
}

// ---------------------------------------------------------------------------
// fc2: [256,1024] x [2,1024]^T + b.
// ---------------------------------------------------------------------------
__global__ void fc2_kernel(const float* __restrict__ act,
                           const float* __restrict__ w,
                           const float* __restrict__ bias,
                           float* __restrict__ out) {
  const int i = blockIdx.x * blockDim.x + threadIdx.x;
  if (i >= 512) return;
  const int n = i & 1;
  const int b = i >> 1;
  float acc = bias[n];
  for (int k = 0; k < 1024; ++k)
    acc += act[b * 1024 + k] * w[n * 1024 + k];
  out[b * 2 + n] = acc;
}

// ---------------------------------------------------------------------------
extern "C" void kernel_launch(void* const* d_in, const int* in_sizes, int n_in,
                              void* d_out, int out_size, void* d_ws, size_t ws_size,
                              hipStream_t stream) {
  const float* image   = (const float*)d_in[0];
  const int*   question= (const int*)  d_in[1];
  const float* stem_w0 = (const float*)d_in[2];
  const float* stem_b0 = (const float*)d_in[3];
  const float* stem_w  = (const float*)d_in[4];
  const float* stem_b  = (const float*)d_in[5];
  const float* exp_w1  = (const float*)d_in[6];
  const float* exp_b1  = (const float*)d_in[7];
  const float* exp_w2  = (const float*)d_in[8];
  const float* exp_b2  = (const float*)d_in[9];
  const float* proj_w  = (const float*)d_in[10];
  const float* proj_b  = (const float*)d_in[11];
  const float* fc1_w   = (const float*)d_in[12];
  const float* fc1_b   = (const float*)d_in[13];
  const float* fc2_w   = (const float*)d_in[14];
  const float* fc2_b   = (const float*)d_in[15];
  float* outp = (float*)d_out;

  float* ws = (float*)d_ws;
  size_t o = 0;
  float* h64   = ws + o; o += (size_t)256 * 64 * 64 * 64;
  float* h32a  = ws + o; o += (size_t)256 * 64 * 32 * 32;
  float* h32b  = ws + o; o += (size_t)256 * 64 * 32 * 32;
  float* h16a  = ws + o; o += (size_t)256 * 64 * 16 * 16;
  float* h16b  = ws + o; o += (size_t)256 * 64 * 16 * 16;
  float* h16c  = ws + o; o += (size_t)256 * 64 * 16 * 16;
  float* projo = ws + o; o += (size_t)256 * 512 * 16 * 16;
  float* pool8 = ws + o; o += (size_t)256 * 512 * 8 * 8;
  float* fc1o  = ws + o; o += (size_t)256 * 1024;

  _Float16* hws = (_Float16*)(ws + o);
  size_t ho = 0;
  _Float16* stemw_pk = hws + ho; ho += (size_t)3  * 9 * 64 * 64;
  _Float16* expw1_pk = hws + ho; ho += (size_t)16 * 9 * 64 * 64;
  _Float16* expw2_pk = hws + ho; ho += (size_t)16 * 9 * 64 * 64;
  _Float16* projw_pk = hws + ho; ho += (size_t)512 * 64;
  _Float16* fc1w_pk  = hws + ho; ho += (size_t)1024 * 32768;

  // one-time weight prepack (~200MB traffic ~ 10us at 23.3 TB/s)
  pack_conv_w_kernel<<<(3 * 36864 + 255) / 256, 256, 0, stream>>>(
      stem_w, stemw_pk, 3 * 36864);
  pack_conv_w_kernel<<<(16 * 36864 + 255) / 256, 256, 0, stream>>>(
      exp_w1, expw1_pk, 16 * 36864);
  pack_conv_w_kernel<<<(16 * 36864 + 255) / 256, 256, 0, stream>>>(
      exp_w2, expw2_pk, 16 * 36864);
  cvt_f16_kernel<<<(32768 + 255) / 256, 256, 0, stream>>>(
      proj_w, projw_pk, 32768);
  cvt_f16_kernel<<<(33554432 + 255) / 256, 256, 0, stream>>>(
      fc1_w, fc1w_pk, 33554432);

  // network
  stem_conv0_kernel<<<(256 * 64 * 64 * 64) / 256, 256, 0, stream>>>(
      image, stem_w0, stem_b0, h64);
  maxpool2_kernel<<<(16777216 + 255) / 256, 256, 0, stream>>>(
      h64, h32a, 64, 64, 16777216);
  conv3x3_wmma_kernel<<<dim3(16, 256), 256, 0, stream>>>(
      h32a, stemw_pk + (size_t)0 * 36864, stem_b + 0 * 64,
      nullptr, -1, nullptr, h32b, 5, 1);
  conv3x3_wmma_kernel<<<dim3(16, 256), 256, 0, stream>>>(
      h32b, stemw_pk + (size_t)1 * 36864, stem_b + 1 * 64,
      nullptr, -1, nullptr, h32a, 5, 1);
  maxpool2_kernel<<<(4194304 + 255) / 256, 256, 0, stream>>>(
      h32a, h16a, 64, 32, 4194304);
  conv3x3_wmma_kernel<<<dim3(4, 256), 256, 0, stream>>>(
      h16a, stemw_pk + (size_t)2 * 36864, stem_b + 2 * 64,
      nullptr, -1, nullptr, h16b, 4, 1);

  float* h = h16b;
  float* other = h16a;
  const int cols[3] = {4, 5, 7};
  for (int t = 0; t < 3; ++t) {
    conv3x3_wmma_kernel<<<dim3(4, 256), 256, 0, stream>>>(
        h, expw1_pk, exp_b1, question, cols[t], nullptr, h16c, 4, 1);
    conv3x3_wmma_kernel<<<dim3(4, 256), 256, 0, stream>>>(
        h16c, expw2_pk, exp_b2, question, cols[t], h, other, 4, 1);
    float* tmp = h; h = other; other = tmp;
  }

  proj_wmma_kernel<<<dim3(1024, 8), 256, 0, stream>>>(h, projw_pk, proj_b, projo);
  maxpool2_kernel<<<(8388608 + 255) / 256, 256, 0, stream>>>(
      projo, pool8, 512, 16, 8388608);
  fc_wmma_kernel<<<dim3(4, 16), 256, 0, stream>>>(
      pool8, fc1w_pk, fc1_b, fc1o, 1024, 32768);
  fc2_kernel<<<2, 256, 0, stream>>>(fc1o, fc2_w, fc2_b, outp);
}
